// LinearDecoder_89000312307893
// MI455X (gfx1250) — compile-verified, matrix operand order checked by reading, not run on hardware
//
#include <hip/hip_runtime.h>
#include <hip/hip_bf16.h>

// ---------------------------------------------------------------------------
// KWTA mask + Linear:  selected = (sel >= kth50(sel_row)) * inp
//                      out      = selected @ W.T + b
// GEMM runs in f16 via V_WMMA_F32_16X16X32_F16 (f32 accumulate); the exact
// fp32 `selected` is emitted directly.
// ---------------------------------------------------------------------------

typedef __attribute__((ext_vector_type(16))) _Float16 v16h;
typedef __attribute__((ext_vector_type(8)))  float    v8f;

#define KPICK 50
#define DDIM  1024

// ---------------- Kernel 1: exact k-th largest per row (radix select) ------
__global__ void kwta_thresh(const float* __restrict__ sel,
                            float* __restrict__ thr) {
    __shared__ unsigned hist[256];
    __shared__ unsigned s_prefix;
    __shared__ unsigned s_k;

    const int row = blockIdx.x;
    const int t   = threadIdx.x;              // 0..255
    const float* p = sel + (size_t)row * DDIM;

    // order-preserving f32 -> u32 mapping (bigger float <-> bigger uint)
    unsigned v[4];
#pragma unroll
    for (int i = 0; i < 4; ++i) {
        unsigned u = __float_as_uint(p[t + 256 * i]);
        v[i] = (u & 0x80000000u) ? ~u : (u | 0x80000000u);
    }

    if (t == 0) { s_prefix = 0u; s_k = KPICK; }
    __syncthreads();

#pragma unroll
    for (int shift = 24; shift >= 0; shift -= 8) {
        hist[t] = 0u;
        __syncthreads();                       // hist cleared, prev prefix visible
        const unsigned prefix = s_prefix;
        const unsigned mask   = (shift == 24) ? 0u : (0xFFFFFFFFu << (shift + 8));
#pragma unroll
        for (int i = 0; i < 4; ++i) {
            if ((v[i] & mask) == prefix)
                atomicAdd(&hist[(v[i] >> shift) & 255u], 1u);
        }
        __syncthreads();
        if (t == 0) {
            unsigned k = s_k, cum = 0u;
            for (int b = 255; b >= 0; --b) {
                const unsigned c = hist[b];
                if (cum + c >= k) {
                    s_k      = k - cum;
                    s_prefix = prefix | ((unsigned)b << shift);
                    break;
                }
                cum += c;
            }
        }
        __syncthreads();
    }

    if (t == 0) {
        const unsigned u = s_prefix;           // k-th largest in mapped domain
        const unsigned f = (u & 0x80000000u) ? (u ^ 0x80000000u) : ~u;
        thr[row] = __uint_as_float(f);
    }
}

// ---------------- Kernel 2: selected = mask*inp (fp32 out + f16 copy) ------
__global__ void select_k(const float* __restrict__ sel,
                         const float* __restrict__ inp,
                         const float* __restrict__ thr,
                         float* __restrict__ out_sel,
                         _Float16* __restrict__ a_h) {
    const size_t i4  = (size_t)blockIdx.x * blockDim.x + threadIdx.x;  // float4 idx
    const size_t idx = i4 * 4;
    const int    row = (int)(idx >> 10);       // D = 1024
    const float  tv  = thr[row];

    const float4 s = *(const float4*)(sel + idx);
    const float4 x = *(const float4*)(inp + idx);
    float4 r;
    r.x = (s.x >= tv) ? x.x : 0.0f;
    r.y = (s.y >= tv) ? x.y : 0.0f;
    r.z = (s.z >= tv) ? x.z : 0.0f;
    r.w = (s.w >= tv) ? x.w : 0.0f;
    *(float4*)(out_sel + idx) = r;

    _Float16 h[4] = { (_Float16)r.x, (_Float16)r.y, (_Float16)r.z, (_Float16)r.w };
    *(uint2*)(a_h + idx) = *(uint2*)h;
}

// ---------------- Kernel 3: W fp32 -> f16 (row-major, no transpose) --------
__global__ void convert_w(const float* __restrict__ W,
                          _Float16* __restrict__ w_h) {
    const size_t idx = ((size_t)blockIdx.x * blockDim.x + threadIdx.x) * 4;
    const float4 w = *(const float4*)(W + idx);
    _Float16 h[4] = { (_Float16)w.x, (_Float16)w.y, (_Float16)w.z, (_Float16)w.w };
    *(uint2*)(w_h + idx) = *(uint2*)h;
}

// ---------------- Kernel 4: WMMA GEMM  out = A @ W^T + b -------------------
// ISA 16-bit fragment layout (wave32): lane = 16*hi + r, r = lane&15.
//   A/B frag: halves 0..7  = K = kb + 8*hi + (0..7)
//             halves 8..15 = K = kb + 16 + 8*hi + (0..7)
//   (two contiguous 16-byte chunks per lane -> two b128 loads)
// C/D 16x16 f32: VGPR j <-> M = 8*hi + j, N = r.
__device__ __forceinline__ v16h load_frag(const _Float16* p) {
    v16h v;
    __builtin_memcpy(&v, p, 16);                                   // K+0..7
    __builtin_memcpy(reinterpret_cast<char*>(&v) + 16, p + 16, 16); // K+16..23
    return v;
}

#define TM 4
#define TN 2

__global__ void wmma_gemm(const _Float16* __restrict__ A,
                          const _Float16* __restrict__ Wh,
                          const float* __restrict__ bias,
                          float* __restrict__ out) {
    const int lane  = threadIdx.x & 31;
    const int wave  = threadIdx.x >> 5;
    const int gwave = blockIdx.x * (blockDim.x >> 5) + wave;

    const int nTilesN = DDIM / (16 * TN);      // 32
    const int mb = gwave / nTilesN;
    const int nb = gwave % nTilesN;
    const int m0 = mb * 16 * TM;
    const int n0 = nb * 16 * TN;

    const int r  = lane & 15;
    const int hi = lane >> 4;

    v8f acc[TM][TN];
#pragma unroll
    for (int i = 0; i < TM; ++i)
#pragma unroll
        for (int j = 0; j < TN; ++j)
            acc[i][j] = (v8f){0.f, 0.f, 0.f, 0.f, 0.f, 0.f, 0.f, 0.f};

    for (int kb = 0; kb < DDIM; kb += 32) {
        v16h bfrag[TN];
#pragma unroll
        for (int tn = 0; tn < TN; ++tn)
            bfrag[tn] = load_frag(Wh + (size_t)(n0 + tn * 16 + r) * DDIM + kb + 8 * hi);
#pragma unroll
        for (int tm = 0; tm < TM; ++tm) {
            const v16h afrag =
                load_frag(A + (size_t)(m0 + tm * 16 + r) * DDIM + kb + 8 * hi);
#pragma unroll
            for (int tn = 0; tn < TN; ++tn) {
                acc[tm][tn] = __builtin_amdgcn_wmma_f32_16x16x32_f16(
                    /*neg_a=*/false, afrag, /*neg_b=*/false, bfrag[tn],
                    /*c_mod=*/(short)0, acc[tm][tn],
                    /*reuse_a=*/false, /*reuse_b=*/false);
            }
        }
    }

#pragma unroll
    for (int tm = 0; tm < TM; ++tm) {
#pragma unroll
        for (int tn = 0; tn < TN; ++tn) {
            const int   col = n0 + tn * 16 + r;
            const float bv  = bias[col];
#pragma unroll
            for (int j = 0; j < 8; ++j) {
                const int rowm = m0 + tm * 16 + hi * 8 + j;
                out[(size_t)rowm * DDIM + col] = acc[tm][tn][j] + bv;
            }
        }
    }
}

// ---------------------------------------------------------------------------
extern "C" void kernel_launch(void* const* d_in, const int* in_sizes, int n_in,
                              void* d_out, int out_size, void* d_ws, size_t ws_size,
                              hipStream_t stream) {
    const float* sel  = (const float*)d_in[0];
    const float* inp  = (const float*)d_in[1];
    const float* W    = (const float*)d_in[2];
    const float* bias = (const float*)d_in[3];

    const int D = DDIM;
    const int B = in_sizes[0] / D;             // 16384

    float* out_sel = (float*)d_out;
    float* out_lin = (float*)d_out + (size_t)B * D;

    char* ws = (char*)d_ws;
    _Float16* a_h = (_Float16*)ws;                                   // B*D*2 bytes
    _Float16* w_h = (_Float16*)(ws + (size_t)B * D * 2);             // D*D*2 bytes
    float*    thr = (float*)(ws + (size_t)B * D * 2 + (size_t)D * D * 2);

    // 1) exact 50th-largest per row
    kwta_thresh<<<B, 256, 0, stream>>>(sel, thr);

    // 2) W -> f16 (row-major == B-fragment order for out = A @ W^T)
    convert_w<<<(D * D / 4) / 256, 256, 0, stream>>>(W, w_h);

    // 3) selected (fp32 out) + f16 copy for the tensor path
    select_k<<<((size_t)B * D / 4) / 256, 256, 0, stream>>>(sel, inp, thr, out_sel, a_h);

    // 4) WMMA GEMM: one wave per 64x32 output tile
    const int waves  = (B / (16 * TM)) * (D / (16 * TN));            // 8192
    const int blocks = waves / 8;                                    // 8 waves/block
    wmma_gemm<<<blocks, 256, 0, stream>>>(a_h, w_h, bias, out_lin);
}
